// Network_77541339562682
// MI455X (gfx1250) — matrix operand order repaired and checked by reading, not compiled, not run on hardware
//
#include <hip/hip_runtime.h>
#include <cstdint>
#include <cstddef>

// ---------------------------------------------------------------- constants
constexpr int NNODES  = 8192;
constexpr int NEDGES  = 163840;
constexpr int NGRAPH  = 32;

#define SQ2F          0.70710678118654752f
#define RINV30        0.18257418583505536f   // 1/sqrt(30)
#define RINV20        0.22360679774997896f   // 1/sqrt(20)
#define SQRT3_RINV20  0.38729833462074170f   // sqrt(3)/sqrt(20)
#define RINV3         0.57735026918962576f   // 1/sqrt(3)
#define RINV10        0.31622776601683794f   // 1/sqrt(10)

typedef __bf16        bf16_t;
typedef __bf16        v16bf __attribute__((ext_vector_type(16)));
typedef float         v8f   __attribute__((ext_vector_type(8)));
typedef unsigned int  u32x4 __attribute__((ext_vector_type(4)));
typedef int           i32x4 __attribute__((ext_vector_type(4)));
typedef int           i32x8 __attribute__((ext_vector_type(8)));
typedef float         f32x4 __attribute__((ext_vector_type(4)));

// ---------------------------------------------------------------- LDS layout (bytes, dynamic smem)
enum : unsigned {
  OFF_IN  = 0,              // [64][64]  bf16 basis tile          8192 B
  OFF_H1  = 8192,           // [64][512] bf16 hidden1            65536 B
  OFF_H2  = 73728,          // [64][512] bf16 hidden2            65536 B
  OFF_W   = 139264,         // weight staging, 2 x 32KB halves   65536 B
  OFF_H3  = 204800,         // [64][64]  bf16 hidden3             8192 B
  OFF_XJ  = 212992,         // [64][<=64] bf16 gathered x_j       8192 B
  OFF_SH  = 221184,         // [64][4]   f32 spherical harmonics  1024 B
  OFF_M   = 222208,         // [64][64]  f32 epilogue accum      16384 B
  OFF_COL = 238592,         // [64]      int dest node             256 B
  SMEM_BYTES = 238848       // ~233 KB of 320 KB WGP LDS
};

// ---------------------------------------------------------------- TDM (Tensor Data Mover) helper
// 2-D (or 1-D with h==1) bf16 tile copy: global -> LDS, tracked by TENSORcnt.
// clang-23 toolchain: 6-arg builtin (g0 v4u, g1 v8i, g2 v4i, g3 v4i, v8i, cpol).
__device__ __forceinline__ void tdm_load_bf16(unsigned lds_off, const void* gsrc,
                                              unsigned w, unsigned h, unsigned stride_elems) {
  unsigned long long ga = (unsigned long long)(uintptr_t)gsrc;
  u32x4 g0;
  g0[0] = 1u;                                                   // count=1, user mode
  g0[1] = lds_off;                                              // lds_addr (bytes)
  g0[2] = (unsigned)ga;                                         // global_addr[31:0]
  g0[3] = ((unsigned)(ga >> 32) & 0x01FFFFFFu) | (2u << 30);    // global_addr[56:32] | type=2
  i32x8 g1;
  g1[0] = (int)(1u << 16);                                      // data_size=1 (2 bytes), mask=0
  g1[1] = (int)((w & 0xFFFFu) << 16);                           // tensor_dim0[15:0]
  g1[2] = (int)(((w >> 16) & 0xFFFFu) | ((h & 0xFFFFu) << 16)); // tensor_dim0[31:16] | tensor_dim1 lo
  g1[3] = (int)((w & 0xFFFFu) << 16);                           // tensor_dim1 hi(=0) | tile_dim0
  g1[4] = (int)(h & 0xFFFFu);                                   // tile_dim1 = h, tile_dim2 = 0
  g1[5] = (int)stride_elems;                                    // tensor_dim0_stride[31:0]
  g1[6] = 0;
  g1[7] = 0;
  i32x4 gz4 = {0, 0, 0, 0};
  i32x8 gz8 = {0, 0, 0, 0, 0, 0, 0, 0};
  __builtin_amdgcn_tensor_load_to_lds(g0, g1, gz4, gz4, gz8, 0);
}

// ---------------------------------------------------------------- WMMA fragment helpers (wave32)
union Frag16 { v16bf v; u32x4 q[2]; };

// A-matrix 16x32 bf16: lane ln=row, hh selects K-half; two contiguous 8-elem runs.
__device__ __forceinline__ v16bf load_afrag(const bf16_t* buf, int ld, int row0, int k0,
                                            int ln, int hh) {
  const bf16_t* p = buf + (row0 + ln) * ld + k0 + 8 * hh;
  Frag16 f;
  f.q[0] = *(const u32x4*)p;
  f.q[1] = *(const u32x4*)(p + 16);
  return f.v;
}
// B-matrix 32x16 bf16: lane%16 = N column, K in table order (8h+s, 16+8h+s-8), strided by ld.
__device__ __forceinline__ v16bf load_bfrag(const bf16_t* buf, int ld, int k0, int n0, int lane) {
  const bf16_t* p = buf + (k0 + 8 * (lane >> 4)) * ld + n0 + (lane & 15);
  v16bf b;
#pragma unroll
  for (int s = 0; s < 8; ++s) {
    b[s] = p[s * ld];
    b[8 + s] = p[(16 + s) * ld];
  }
  return b;
}

__device__ __forceinline__ v8f wmma_bf16(v16bf a, v16bf b, v8f c) {
  return __builtin_amdgcn_wmma_f32_16x16x32_bf16(false, a, false, b, (short)0, c, false, false);
}

__device__ __forceinline__ v8f zero8() {
  v8f z = {0.f, 0.f, 0.f, 0.f, 0.f, 0.f, 0.f, 0.f};
  return z;
}

// Fast transcendental epilogues: v_exp_f32 + v_rcp_f32 only (no IEEE div sequence).
__device__ __forceinline__ float fast_sigmoid(float x) {
  return __builtin_amdgcn_rcpf(1.f + __expf(-x));
}
__device__ __forceinline__ float swishf(float x) {
  return x * fast_sigmoid(x);
}

// ---------------------------------------------------------------- fused edge-MLP + tensor-product conv
// One workgroup = 64 edges, 8 waves (wave32). Full 40(->64)-500(->512)-500-50(->64)-Dout
// MLP chained through LDS; weights streamed by TDM, double buffered; epilogue fuses
// the e3nn tensor product and the segment-sum scatter.
template <int CONV>
__global__ __launch_bounds__(256, 1)
void conv_mlp_kernel(const bf16_t* __restrict__ B0, const float* __restrict__ shg,
                     const int* __restrict__ eidx, const bf16_t* __restrict__ Xsrc,
                     const bf16_t* __restrict__ W1, const bf16_t* __restrict__ W2,
                     const bf16_t* __restrict__ W3, const bf16_t* __restrict__ W4,
                     const float* __restrict__ b1, const float* __restrict__ b2,
                     const float* __restrict__ b3, const float* __restrict__ b4,
                     float* __restrict__ agg) {
  constexpr int XLD = (CONV == 1) ? 32 : 64;
  extern __shared__ unsigned char smem[];
  bf16_t* sIn = (bf16_t*)(smem + OFF_IN);
  bf16_t* sH1 = (bf16_t*)(smem + OFF_H1);
  bf16_t* sH2 = (bf16_t*)(smem + OFF_H2);
  bf16_t* sW  = (bf16_t*)(smem + OFF_W);
  bf16_t* sH3 = (bf16_t*)(smem + OFF_H3);
  bf16_t* sXj = (bf16_t*)(smem + OFF_XJ);
  float*  sSh = (float*)(smem + OFF_SH);
  float*  sM  = (float*)(smem + OFF_M);
  int*    sCol = (int*)(smem + OFF_COL);

  const int tid  = threadIdx.x;
  const int wid  = tid >> 5;
  const int lane = tid & 31;
  const int rt   = wid & 3;    // row-tile (16 edges each)
  const int ch   = wid >> 2;   // column half
  const int ln   = lane & 15;
  const int hh   = lane >> 4;
  const int eb   = blockIdx.x * 64;

  // ---- stage 0: async-stage basis + layer-1 weights via TDM, gather the rest ----
  if (wid == 0) {
    tdm_load_bf16(OFF_IN, B0 + eb * 64, 64 * 64, 1, 64 * 64);
    tdm_load_bf16(OFF_W, W1, 64 * 512, 1, 64 * 512);
  }
  for (int i = tid; i < 64; i += 256) {
    sCol[i] = eidx[NEDGES + eb + i];
    *(f32x4*)(sSh + i * 4) = *(const f32x4*)(shg + (eb + i) * 4);
  }
  for (int i = tid; i < 64 * (XLD / 8); i += 256) {
    int e = i / (XLD / 8), c8 = i % (XLD / 8);
    int sn = eidx[eb + e];
    *(u32x4*)(sXj + e * XLD + c8 * 8) = *(const u32x4*)(Xsrc + sn * XLD + c8 * 8);
  }
  if (wid == 0) __builtin_amdgcn_s_wait_tensorcnt((short)0);
  __syncthreads();

  // ---- CONV1: seed epilogue accumulator with bias (.) x_j contraction ----
  if (CONV == 1) {
    for (int i = tid; i < 64 * 50; i += 256) {
      int e = i / 50, o = i % 50;
      float acc = 0.f;
      if (o < 40) {
        for (int u = 0; u < 30; ++u) acc += (float)sXj[e * XLD + u] * b4[u * 40 + o];
        sM[e * 64 + o] = acc;
      } else {
        int p = o - 40;
        for (int u = 0; u < 30; ++u) acc += (float)sXj[e * XLD + u] * b4[1200 + u * 10 + p];
        sM[e * 64 + 40 + p] = acc;
      }
    }
  }

  // ---- layer 1: H1 = swish(IN[64x64] @ W1[64x512] + b1) ----
  for (int nt = 0; nt < 16; ++nt) {
    const int n0 = ch * 256 + nt * 16;
    v8f acc = zero8();
#pragma unroll
    for (int kc = 0; kc < 2; ++kc) {
      v16bf a = load_afrag(sIn, 64, rt * 16, kc * 32, ln, hh);
      v16bf b = load_bfrag(sW, 512, kc * 32, n0, lane);
      acc = wmma_bf16(a, b, acc);
    }
    const float bias = b1[n0 + ln];
#pragma unroll
    for (int r = 0; r < 8; ++r) {
      float x = swishf(acc[r] + bias);
      sH1[(rt * 16 + r + 8 * hh) * 512 + n0 + ln] = (bf16_t)x;
    }
  }
  __syncthreads();

  // ---- layer 2: H2 = swish(H1 @ W2[512x512] + b2); TDM double-buffered K-chunks ----
  if (wid == 0) tdm_load_bf16(OFF_W, W2, 32 * 512, 1, 32 * 512);
  v8f acc2[16];
#pragma unroll
  for (int t = 0; t < 16; ++t) acc2[t] = zero8();
  for (int c = 0; c < 16; ++c) {
    if (wid == 0) {
      if (c + 1 < 16) {
        tdm_load_bf16(OFF_W + (unsigned)(((c + 1) & 1) * 32768), W2 + (c + 1) * 32 * 512,
                      32 * 512, 1, 32 * 512);
        __builtin_amdgcn_s_wait_tensorcnt((short)1);
      } else {
        __builtin_amdgcn_s_wait_tensorcnt((short)0);
      }
    }
    __syncthreads();
    const bf16_t* wb = sW + (c & 1) * 16384;
    v16bf a = load_afrag(sH1, 512, rt * 16, c * 32, ln, hh);
#pragma unroll
    for (int nt = 0; nt < 16; ++nt) {
      v16bf b = load_bfrag(wb, 512, 0, ch * 256 + nt * 16, lane);
      acc2[nt] = wmma_bf16(a, b, acc2[nt]);
    }
    __syncthreads();
  }
  for (int nt = 0; nt < 16; ++nt) {
    const int n0 = ch * 256 + nt * 16;
    const float bias = b2[n0 + ln];
#pragma unroll
    for (int r = 0; r < 8; ++r) {
      float x = swishf(acc2[nt][r] + bias);
      sH2[(rt * 16 + r + 8 * hh) * 512 + n0 + ln] = (bf16_t)x;
    }
  }
  __syncthreads();

  // ---- layer 3: H3 = swish(H2 @ W3[512x64] + b3) ----
  if (wid == 0) {
    tdm_load_bf16(OFF_W, W3, 512 * 64, 1, 512 * 64);
    __builtin_amdgcn_s_wait_tensorcnt((short)0);
  }
  __syncthreads();
  {
    v8f acc3[2];
    acc3[0] = zero8();
    acc3[1] = zero8();
    for (int kc = 0; kc < 16; ++kc) {
      v16bf a = load_afrag(sH2, 512, rt * 16, kc * 32, ln, hh);
#pragma unroll
      for (int t = 0; t < 2; ++t) {
        v16bf b = load_bfrag(sW, 64, kc * 32, (ch * 2 + t) * 16, lane);
        acc3[t] = wmma_bf16(a, b, acc3[t]);
      }
    }
#pragma unroll
    for (int t = 0; t < 2; ++t) {
      const int n0 = (ch * 2 + t) * 16;
      const float bias = b3[n0 + ln];
#pragma unroll
      for (int r = 0; r < 8; ++r) {
        float x = swishf(acc3[t][r] + bias);
        sH3[(rt * 16 + r + 8 * hh) * 64 + n0 + ln] = (bf16_t)x;
      }
    }
  }
  __syncthreads();

  if (CONV == 1) {
    // ---- layer 4 + TP: w = H3 @ W4[64x1536]; fused contraction with x_j, never hits HBM ----
    if (wid == 0) tdm_load_bf16(OFF_W, W4, 256, 64, 1536);
    for (int c = 0; c < 6; ++c) {
      if (wid == 0) {
        if (c + 1 < 6) {
          tdm_load_bf16(OFF_W + (unsigned)(((c + 1) & 1) * 32768), W4 + (c + 1) * 256,
                        256, 64, 1536);
          __builtin_amdgcn_s_wait_tensorcnt((short)1);
        } else {
          __builtin_amdgcn_s_wait_tensorcnt((short)0);
        }
      }
      __syncthreads();
      const bf16_t* wb = sW + (c & 1) * 16384;
      for (int t = 0; t < 8; ++t) {
        v8f acc = zero8();
#pragma unroll
        for (int kc = 0; kc < 2; ++kc) {
          v16bf a = load_afrag(sH3, 64, rt * 16, kc * 32, ln, hh);
          v16bf b = load_bfrag(wb, 256, kc * 32, (ch * 8 + t) * 16, lane);
          acc = wmma_bf16(a, b, acc);
        }
        const int jg = c * 256 + (ch * 8 + t) * 16 + ln;
        if (jg < 1200) {
          const int u = jg / 40, v = jg % 40;
#pragma unroll
          for (int r = 0; r < 8; ++r) {
            const int e = rt * 16 + r + 8 * hh;
            atomicAdd(&sM[e * 64 + v], (float)sXj[e * XLD + u] * acc[r]);
          }
        } else if (jg < 1500) {
          const int q = jg - 1200, u = q / 10, p = q % 10;
#pragma unroll
          for (int r = 0; r < 8; ++r) {
            const int e = rt * 16 + r + 8 * hh;
            atomicAdd(&sM[e * 64 + 40 + p], (float)sXj[e * XLD + u] * acc[r]);
          }
        }
      }
      __syncthreads();
    }
    // scatter messages (x sh, /sqrt(30)) into node aggregation [N,72]
    for (int i = tid; i < 64 * 70; i += 256) {
      const int e = i / 70, f = i % 70;
      const int cn = sCol[e];
      float msg;
      if (f < 40) {
        msg = sM[e * 64 + f] * sSh[e * 4] * RINV30;
      } else {
        const int f2 = f - 40, p = f2 / 3, d = f2 % 3;
        msg = sM[e * 64 + 40 + p] * sSh[e * 4 + 1 + d] * RINV30;
      }
      atomicAdd(&agg[cn * 72 + f], msg);
    }
  } else {
    // ---- CONV2 layer 4: w2 = H3 @ W4[64x64] + b4; per-edge CG contraction, scalar scatter ----
    if (wid == 0) {
      tdm_load_bf16(OFF_W, W4, 64 * 64, 1, 64 * 64);
      __builtin_amdgcn_s_wait_tensorcnt((short)0);
    }
    __syncthreads();
#pragma unroll
    for (int t = 0; t < 2; ++t) {
      v8f acc = zero8();
#pragma unroll
      for (int kc = 0; kc < 2; ++kc) {
        v16bf a = load_afrag(sH3, 64, rt * 16, kc * 32, ln, hh);
        v16bf b = load_bfrag(sW, 64, kc * 32, (ch * 2 + t) * 16, lane);
        acc = wmma_bf16(a, b, acc);
      }
      const int jg = (ch * 2 + t) * 16 + ln;
      const float bias = b4[jg];
#pragma unroll
      for (int r = 0; r < 8; ++r) {
        const int e = rt * 16 + r + 8 * hh;
        sM[e * 64 + jg] = acc[r] + bias;
      }
    }
    __syncthreads();
    if (tid < 64) {
      const int e = tid, cn = sCol[e];
      const float sh0 = sSh[e * 4], s1v = sSh[e * 4 + 1], s2v = sSh[e * 4 + 2],
                  s3v = sSh[e * 4 + 3];
      float m0a = 0.f;
      for (int u = 0; u < 30; ++u) m0a += (float)sXj[e * XLD + u] * sM[e * 64 + u];
      m0a *= sh0 * RINV30;
      float m0b = 0.f;
      for (int u = 0; u < 10; ++u) {
        float dot = ((float)sXj[e * XLD + 30 + u * 3] * s1v +
                     (float)sXj[e * XLD + 31 + u * 3] * s2v +
                     (float)sXj[e * XLD + 32 + u * 3] * s3v) * RINV3;
        m0b += dot * sM[e * 64 + 30 + u];
      }
      m0b *= RINV10;
      atomicAdd(&agg[cn], m0a + m0b);
    }
  }
}

// ---------------------------------------------------------------- side kernels
__global__ void zero_kernel(float* agg1, float* agg2, float* out) {
  int i = blockIdx.x * blockDim.x + threadIdx.x;
  if (i < NNODES * 72) agg1[i] = 0.f;
  if (i < NNODES) agg2[i] = 0.f;
  if (i < NGRAPH) out[i] = 0.f;
}

__global__ void pack_w_kernel(const float* __restrict__ src, bf16_t* __restrict__ dst,
                              int r, int c, int pr, int pc) {
  int i = blockIdx.x * blockDim.x + threadIdx.x;
  if (i >= pr * pc) return;
  int rr = i / pc, cc = i % pc;
  float v = (rr < r && cc < c) ? src[rr * c + cc] : 0.f;
  dst[i] = (bf16_t)v;
}

__global__ void pack_b_kernel(const float* __restrict__ src, float* __restrict__ dst,
                              int n, int pn) {
  int i = blockIdx.x * blockDim.x + threadIdx.x;
  if (i >= pn) return;
  dst[i] = (i < n) ? src[i] : 0.f;
}

__global__ void node_prep1_kernel(const int* __restrict__ z, const float* __restrict__ emb,
                                  const float* __restrict__ si1, bf16_t* __restrict__ hn,
                                  float* __restrict__ s1) {
  int n = blockIdx.x * blockDim.x + threadIdx.x;
  if (n >= NNODES) return;
  float h[30];
  const float* er = emb + z[n] * 30;
#pragma unroll
  for (int i = 0; i < 30; ++i) {
    h[i] = er[i];
    hn[n * 32 + i] = (bf16_t)h[i];
  }
  hn[n * 32 + 30] = (bf16_t)0.f;
  hn[n * 32 + 31] = (bf16_t)0.f;
  for (int j = 0; j < 40; ++j) {
    float acc = 0.f;
#pragma unroll
    for (int i = 0; i < 30; ++i) acc += h[i] * si1[i * 40 + j];
    s1[n * 40 + j] = acc * RINV30;
  }
}

__global__ void edge_prep_kernel(const float* __restrict__ pos, const int* __restrict__ eidx,
                                 float* __restrict__ shg, bf16_t* __restrict__ B0) {
  int e = blockIdx.x * blockDim.x + threadIdx.x;
  if (e >= NEDGES) return;
  const int a = eidx[e], b = eidx[NEDGES + e];
  const float dx = pos[a * 3 + 0] - pos[b * 3 + 0];
  const float dy = pos[a * 3 + 1] - pos[b * 3 + 1];
  const float dz = pos[a * 3 + 2] - pos[b * 3 + 2];
  const float rr = sqrtf(dx * dx + dy * dy + dz * dz + 1e-12f);
  const float inv = __builtin_amdgcn_rcpf(rr);
  shg[e * 4 + 0] = RINV20;
  shg[e * 4 + 1] = SQRT3_RINV20 * dx * inv;
  shg[e * 4 + 2] = SQRT3_RINV20 * dy * inv;
  shg[e * 4 + 3] = SQRT3_RINV20 * dz * inv;
  const float winv = 3.9f;                       // 39/10 = 1/width
#pragma unroll
  for (int k = 0; k < 40; ++k) {
    float t = rr * winv - (float)k;
    B0[e * 64 + k] = (bf16_t)__expf(-t * t);
  }
#pragma unroll
  for (int k = 40; k < 64; ++k) B0[e * 64 + k] = (bf16_t)0.f;
}

__global__ void node_gate_kernel(const float* __restrict__ s1, const float* __restrict__ agg1,
                                 const float* __restrict__ si2, bf16_t* __restrict__ h2,
                                 float* __restrict__ s2) {
  int n = blockIdx.x * blockDim.x + threadIdx.x;
  if (n >= NNODES) return;
  float s2acc = 0.f;
#pragma unroll
  for (int f = 0; f < 30; ++f) {
    float x = SQ2F * (s1[n * 40 + f] + agg1[n * 72 + f]);
    float sw = swishf(x);
    h2[n * 64 + f] = (bf16_t)sw;
    s2acc += sw * si2[f];
  }
#pragma unroll
  for (int p = 0; p < 10; ++p) {
    float gx = SQ2F * (s1[n * 40 + 30 + p] + agg1[n * 72 + 30 + p]);
    float g = fast_sigmoid(gx);
#pragma unroll
    for (int d = 0; d < 3; ++d)
      h2[n * 64 + 30 + p * 3 + d] = (bf16_t)(agg1[n * 72 + 40 + p * 3 + d] * g);
  }
#pragma unroll
  for (int k = 60; k < 64; ++k) h2[n * 64 + k] = (bf16_t)0.f;
  s2[n] = s2acc * RINV30;
}

__global__ void finalize_kernel(const float* __restrict__ s2, const float* __restrict__ agg2,
                                const int* __restrict__ batch, float* __restrict__ out) {
  int n = blockIdx.x * blockDim.x + threadIdx.x;
  if (n >= NNODES) return;
  float hf0 = SQ2F * (s2[n] + agg2[n]);
  atomicAdd(&out[batch[n]], hf0);
}

// ---------------------------------------------------------------- workspace layout (bytes)
enum : size_t {
  OW1A = 0,
  OW2A = OW1A + (size_t)64 * 512 * 2,
  OW3A = OW2A + (size_t)512 * 512 * 2,
  OW4A = OW3A + (size_t)512 * 64 * 2,
  OW1B = OW4A + (size_t)64 * 1536 * 2,
  OW2B = OW1B + (size_t)64 * 512 * 2,
  OW3B = OW2B + (size_t)512 * 512 * 2,
  OW4B = OW3B + (size_t)512 * 64 * 2,
  OB1A = OW4B + (size_t)64 * 64 * 2,
  OB2A = OB1A + 512 * 4,
  OB3A = OB2A + 512 * 4,
  OB4A = OB3A + 64 * 4,
  OB1B = OB4A + 1536 * 4,
  OB2B = OB1B + 512 * 4,
  OB3B = OB2B + 512 * 4,
  OB4B = OB3B + 64 * 4,
  OB0  = OB4B + 64 * 4,
  OSH  = OB0 + (size_t)NEDGES * 64 * 2,
  OHN  = OSH + (size_t)NEDGES * 4 * 4,
  OS1  = OHN + (size_t)NNODES * 32 * 2,
  OAG1 = OS1 + (size_t)NNODES * 40 * 4,
  OH2  = OAG1 + (size_t)NNODES * 72 * 4,
  OS2  = OH2 + (size_t)NNODES * 64 * 2,
  OAG2 = OS2 + (size_t)NNODES * 4,
  OEND = OAG2 + (size_t)NNODES * 4
};

// ---------------------------------------------------------------- launch
extern "C" void kernel_launch(void* const* d_in, const int* in_sizes, int n_in,
                              void* d_out, int out_size, void* d_ws, size_t ws_size,
                              hipStream_t stream) {
  (void)in_sizes; (void)n_in; (void)out_size; (void)ws_size;
  const float* pos = (const float*)d_in[0];
  const float* emb = (const float*)d_in[1];
  const float* si1 = (const float*)d_in[2];
  const float* si2 = (const float*)d_in[3];
  const float* rm[16];
  for (int i = 0; i < 16; ++i) rm[i] = (const float*)d_in[4 + i];
  const int* z     = (const int*)d_in[20];
  const int* eidx  = (const int*)d_in[21];
  const int* batch = (const int*)d_in[22];
  float* out = (float*)d_out;
  char* ws = (char*)d_ws;

  bf16_t* W1a = (bf16_t*)(ws + OW1A);
  bf16_t* W2a = (bf16_t*)(ws + OW2A);
  bf16_t* W3a = (bf16_t*)(ws + OW3A);
  bf16_t* W4a = (bf16_t*)(ws + OW4A);
  bf16_t* W1b = (bf16_t*)(ws + OW1B);
  bf16_t* W2b = (bf16_t*)(ws + OW2B);
  bf16_t* W3b = (bf16_t*)(ws + OW3B);
  bf16_t* W4b = (bf16_t*)(ws + OW4B);
  float* b1a = (float*)(ws + OB1A);
  float* b2a = (float*)(ws + OB2A);
  float* b3a = (float*)(ws + OB3A);
  float* b4a = (float*)(ws + OB4A);
  float* b1b = (float*)(ws + OB1B);
  float* b2b = (float*)(ws + OB2B);
  float* b3b = (float*)(ws + OB3B);
  float* b4b = (float*)(ws + OB4B);
  bf16_t* B0 = (bf16_t*)(ws + OB0);
  float* SH  = (float*)(ws + OSH);
  bf16_t* HN = (bf16_t*)(ws + OHN);
  float* S1  = (float*)(ws + OS1);
  float* AG1 = (float*)(ws + OAG1);
  bf16_t* H2 = (bf16_t*)(ws + OH2);
  float* S2  = (float*)(ws + OS2);
  float* AG2 = (float*)(ws + OAG2);

  zero_kernel<<<(NNODES * 72 + 255) / 256, 256, 0, stream>>>(AG1, AG2, out);

  // repack weights f32 -> padded bf16 (row-major [K][N])
  pack_w_kernel<<<(64 * 512 + 255) / 256, 256, 0, stream>>>(rm[0], W1a, 40, 500, 64, 512);
  pack_w_kernel<<<(512 * 512 + 255) / 256, 256, 0, stream>>>(rm[2], W2a, 500, 500, 512, 512);
  pack_w_kernel<<<(512 * 64 + 255) / 256, 256, 0, stream>>>(rm[4], W3a, 500, 50, 512, 64);
  pack_w_kernel<<<(64 * 1536 + 255) / 256, 256, 0, stream>>>(rm[6], W4a, 50, 1500, 64, 1536);
  pack_w_kernel<<<(64 * 512 + 255) / 256, 256, 0, stream>>>(rm[8], W1b, 40, 500, 64, 512);
  pack_w_kernel<<<(512 * 512 + 255) / 256, 256, 0, stream>>>(rm[10], W2b, 500, 500, 512, 512);
  pack_w_kernel<<<(512 * 64 + 255) / 256, 256, 0, stream>>>(rm[12], W3b, 500, 50, 512, 64);
  pack_w_kernel<<<(64 * 64 + 255) / 256, 256, 0, stream>>>(rm[14], W4b, 50, 40, 64, 64);
  pack_b_kernel<<<2, 256, 0, stream>>>(rm[1], b1a, 500, 512);
  pack_b_kernel<<<2, 256, 0, stream>>>(rm[3], b2a, 500, 512);
  pack_b_kernel<<<1, 64, 0, stream>>>(rm[5], b3a, 50, 64);
  pack_b_kernel<<<6, 256, 0, stream>>>(rm[7], b4a, 1500, 1536);
  pack_b_kernel<<<2, 256, 0, stream>>>(rm[9], b1b, 500, 512);
  pack_b_kernel<<<2, 256, 0, stream>>>(rm[11], b2b, 500, 512);
  pack_b_kernel<<<1, 64, 0, stream>>>(rm[13], b3b, 50, 64);
  pack_b_kernel<<<1, 64, 0, stream>>>(rm[15], b4b, 40, 64);

  node_prep1_kernel<<<(NNODES + 255) / 256, 256, 0, stream>>>(z, emb, si1, HN, S1);
  edge_prep_kernel<<<(NEDGES + 255) / 256, 256, 0, stream>>>(pos, eidx, SH, B0);

  conv_mlp_kernel<1><<<NEDGES / 64, 256, SMEM_BYTES, stream>>>(
      B0, SH, eidx, HN, W1a, W2a, W3a, W4a, b1a, b2a, b3a, b4a, AG1);

  node_gate_kernel<<<(NNODES + 255) / 256, 256, 0, stream>>>(S1, AG1, si2, H2, S2);

  conv_mlp_kernel<2><<<NEDGES / 64, 256, SMEM_BYTES, stream>>>(
      B0, SH, eidx, H2, W1b, W2b, W3b, W4b, b1b, b2b, b3b, b4b, AG2);

  finalize_kernel<<<(NNODES + 255) / 256, 256, 0, stream>>>(S2, AG2, batch, out);
}
